// GATSAGE_33638183862592
// MI455X (gfx1250) — matrix operand order repaired and checked by reading, not compiled, not run on hardware
//
#include <hip/hip_runtime.h>

// ---------------------------------------------------------------------------
// Types for CDNA5 WMMA (wave32, v_wmma_f32_16x16x32_bf16)
// ---------------------------------------------------------------------------
typedef __attribute__((ext_vector_type(16))) __bf16 v16bf;
typedef __attribute__((ext_vector_type(8)))  __bf16 v8bf;
typedef __attribute__((ext_vector_type(8)))  float  v8f;

#define GN    100000
#define GE    3200000
#define GINF  512
#define GHID  256
#define GHEADS 8
#define GNCLS 16

#define CEILDIV(a,b) (((a)+(b)-1)/(b))

// ---------------------------------------------------------------------------
// Small helper kernels
// ---------------------------------------------------------------------------
__global__ void k_cast_bf16(const float* __restrict__ s, __bf16* __restrict__ d, long n) {
    long i = (long)blockIdx.x * blockDim.x + threadIdx.x;
    if (i < n) d[i] = (__bf16)s[i];
}

__global__ void k_fill_f32(float* __restrict__ p, float v, long n) {
    long i = (long)blockIdx.x * blockDim.x + threadIdx.x;
    if (i < n) p[i] = v;
}

__global__ void k_fill_u32(unsigned* __restrict__ p, unsigned v, long n) {
    long i = (long)blockIdx.x * blockDim.x + threadIdx.x;
    if (i < n) p[i] = v;
}

__global__ void k_degree(const int* __restrict__ dst, float* __restrict__ deg, int E) {
    long i = (long)blockIdx.x * blockDim.x + threadIdx.x;
    if (i < E) atomicAdd(&deg[dst[i]], 1.0f);
}

// acc[dst] += h[src] over all edges; 4 floats per thread
__global__ void k_scatter_add(const float* __restrict__ h, const int* __restrict__ src,
                              const int* __restrict__ dst, float* __restrict__ acc,
                              int E, int F) {
    long i = (long)blockIdx.x * blockDim.x + threadIdx.x;
    long total = (long)E * (F >> 2);
    if (i >= total) return;
    int chunks = F >> 2;
    long e = i / chunks;
    int  c = (int)(i - e * chunks) << 2;
    int  s = src[e], d = dst[e];
    float4 v = *(const float4*)(h + (size_t)s * F + c);
    float* p = acc + (size_t)d * F + c;
    atomicAdd(p + 0, v.x); atomicAdd(p + 1, v.y);
    atomicAdd(p + 2, v.z); atomicAdd(p + 3, v.w);
}

__global__ void k_div_deg(float* __restrict__ acc, const float* __restrict__ deg, long n, int F) {
    long i = (long)blockIdx.x * blockDim.x + threadIdx.x;
    if (i < n) acc[i] = acc[i] / fmaxf(deg[i / F], 1.0f);
}

// ---------------------------------------------------------------------------
// GAT edge kernels
// ---------------------------------------------------------------------------
__device__ inline void edge_sd(long e, const int* src, const int* dst, int E, int& s, int& d) {
    if (e < E) { s = src[e]; d = dst[e]; }
    else       { s = d = (int)(e - E); }   // appended self loops
}

__device__ inline float leaky02(float v) { return v < 0.f ? 0.2f * v : v; }

// monotonic float <-> uint encoding for atomicMax on floats
__device__ inline unsigned encf(float f) {
    unsigned u = __float_as_uint(f);
    return (u & 0x80000000u) ? ~u : (u | 0x80000000u);
}
__device__ inline float decf(unsigned u) {
    return (u & 0x80000000u) ? __uint_as_float(u & 0x7FFFFFFFu) : __uint_as_float(~u);
}

// el/er: per (node, head) dot of feat row with attn vectors; one wave per head
__global__ __launch_bounds__(256) void k_el_er(const float* __restrict__ feat,
                                               const float* __restrict__ attn_l,
                                               const float* __restrict__ attn_r,
                                               float* __restrict__ el, float* __restrict__ er) {
    int n = blockIdx.x;
    int hw = threadIdx.x >> 5;          // head == wave index (8 waves)
    int lane = threadIdx.x & 31;
    const float* f  = feat + (size_t)n * (GHEADS * GHID) + (size_t)hw * GHID + lane * 8;
    const float* al = attn_l + (size_t)hw * GHID + lane * 8;
    const float* ar = attn_r + (size_t)hw * GHID + lane * 8;
    float sl = 0.f, sr = 0.f;
    #pragma unroll
    for (int q = 0; q < 2; ++q) {       // 8 contiguous floats per lane
        float4 x = *(const float4*)(f + q * 4);
        float4 a = *(const float4*)(al + q * 4);
        float4 b = *(const float4*)(ar + q * 4);
        sl += x.x * a.x + x.y * a.y + x.z * a.z + x.w * a.w;
        sr += x.x * b.x + x.y * b.y + x.z * b.z + x.w * b.w;
    }
    #pragma unroll
    for (int o = 16; o > 0; o >>= 1) {   // wave32 butterfly reduction
        sl += __shfl_xor(sl, o, 32);
        sr += __shfl_xor(sr, o, 32);
    }
    if (lane == 0) {
        el[(size_t)n * GHEADS + hw] = sl;
        er[(size_t)n * GHEADS + hw] = sr;
    }
}

__global__ void k_edge_max(const int* __restrict__ src, const int* __restrict__ dst,
                           const float* __restrict__ el, const float* __restrict__ er,
                           unsigned* __restrict__ emax, int E, int Nn) {
    long i = (long)blockIdx.x * blockDim.x + threadIdx.x;
    long total = (long)(E + Nn) * GHEADS;
    if (i >= total) return;
    int  h = (int)(i & (GHEADS - 1));
    long e = i >> 3;
    int s, d; edge_sd(e, src, dst, E, s, d);
    float v = leaky02(el[(size_t)s * GHEADS + h] + er[(size_t)d * GHEADS + h]);
    atomicMax(&emax[(size_t)d * GHEADS + h], encf(v));
}

__global__ void k_edge_exp(const int* __restrict__ src, const int* __restrict__ dst,
                           const float* __restrict__ el, const float* __restrict__ er,
                           const unsigned* __restrict__ emax, float* __restrict__ denom,
                           float* __restrict__ exb, int E, int Nn) {
    long i = (long)blockIdx.x * blockDim.x + threadIdx.x;
    long total = (long)(E + Nn) * GHEADS;
    if (i >= total) return;
    int  h = (int)(i & (GHEADS - 1));
    long e = i >> 3;
    int s, d; edge_sd(e, src, dst, E, s, d);
    float v  = leaky02(el[(size_t)s * GHEADS + h] + er[(size_t)d * GHEADS + h]);
    float ex = __expf(v - decf(emax[(size_t)d * GHEADS + h]));
    exb[(size_t)e * GHEADS + h] = ex;
    atomicAdd(&denom[(size_t)d * GHEADS + h], ex);
}

// gat[dst] += alpha * feat[src]; one thread per (edge, 16-float chunk of 2048)
__global__ void k_edge_msg(const int* __restrict__ src, const int* __restrict__ dst,
                           const float* __restrict__ feat, const float* __restrict__ exb,
                           const float* __restrict__ denom, float* __restrict__ gat,
                           int E, int Nn) {
    long i = (long)blockIdx.x * blockDim.x + threadIdx.x;
    long total = (long)(E + Nn) * 128;
    if (i >= total) return;
    int  c = (int)(i & 127);             // 16-float chunk index within 2048
    long e = i >> 7;
    int  h = c >> 4;                     // head = chunk*16 / 256
    int s, d; edge_sd(e, src, dst, E, s, d);
    float a = exb[(size_t)e * GHEADS + h] / denom[(size_t)d * GHEADS + h];
    const float4* fp = (const float4*)(feat + (size_t)s * (GHEADS * GHID) + (size_t)c * 16);
    float* gp = gat + (size_t)d * (GHEADS * GHID) + (size_t)c * 16;
    #pragma unroll
    for (int j = 0; j < 4; ++j) {
        float4 v = fp[j];
        atomicAdd(gp + j * 4 + 0, v.x * a);
        atomicAdd(gp + j * 4 + 1, v.y * a);
        atomicAdd(gp + j * 4 + 2, v.z * a);
        atomicAdd(gp + j * 4 + 3, v.w * a);
    }
}

// ---------------------------------------------------------------------------
// bf16 WMMA GEMM: D[M,N] = A[M,K] @ B[K,N] (+add0 +add1 +bias, optional ReLU)
// Block: 256 threads = 8 waves (4x2), tile 128x64, BK=32, double-buffered LDS.
// A tile uses CDNA5 async global->LDS DMA (ASYNCcnt); B tile is staged through
// VGPRs and stored transposed so fragment reads are contiguous ds_load_b128.
// ---------------------------------------------------------------------------
#define BM 128
#define BN 64
#define BK 32
#define LDA_S 40   // padded LDS row stride (elems) for A tile
#define LDB_S 40   // padded LDS row stride (elems) for Bt tile

__device__ inline v8bf v8bf_zero() {
    v8bf r;
    #pragma unroll
    for (int i = 0; i < 8; ++i) r[i] = (__bf16)0.0f;
    return r;
}
__device__ inline v16bf cat8(v8bf lo, v8bf hi) {
    v16bf r;
    #pragma unroll
    for (int i = 0; i < 8; ++i) { r[i] = lo[i]; r[i + 8] = hi[i]; }
    return r;
}

__global__ __launch_bounds__(256) void k_gemm_bf16(
    const __bf16* __restrict__ A, const __bf16* __restrict__ B,
    float* __restrict__ D,
    const float* __restrict__ add0, const float* __restrict__ add1,
    const float* __restrict__ bias,
    int M, int N, int K, int doRelu)
{
    __shared__ __bf16 sA[2][BM * LDA_S];
    __shared__ __bf16 sB[2][BN * LDB_S];

    const int tid  = threadIdx.x;
    const int lane = tid & 31;
    const int wv   = tid >> 5;
    const int wm   = wv >> 1;            // 0..3 (M direction)
    const int wn   = wv & 1;             // 0..1 (N direction)
    const int l15  = lane & 15;
    const int lh   = lane >> 4;          // lane-half: 0 or 1
    const int tileM = blockIdx.x * BM;
    const int tileN = blockIdx.y * BN;

    // global->LDS staging assignments
    const int arow = tid >> 1, acol = (tid & 1) * 16;     // A: 128 rows x 32 cols
    const int brow = tid >> 3, bcol = (tid & 7) * 8;      // B: 32 rows x 64 cols

    // clamped A row so async issue is convergent; OOB rows only feed masked outputs
    const int arowc = (tileM + arow < M) ? (tileM + arow) : (M - 1);
    const __bf16* aBase = A + (size_t)arowc * K + acol;

    // A tile: async DMA global->LDS, two b128 per thread (row chunks are
    // contiguous in both global and padded LDS layout: +0 and +16 bytes)
    auto asyncA = [&](int k0, int buf) {
        unsigned lds = (unsigned)(size_t)&sA[buf][arow * LDA_S + acol];
        unsigned long long ga = (unsigned long long)(aBase + k0);
        asm volatile(
            "global_load_async_to_lds_b128 %0, %1, off\n\t"
            "global_load_async_to_lds_b128 %0, %1, off offset:16"
            :: "v"(lds), "v"(ga) : "memory");
    };
    auto waitAsync = [&]() {
        asm volatile("s_wait_asynccnt 0x0" ::: "memory");
    };

    v8bf rb;
    auto stageB = [&](int k0) {
        int gc = tileN + bcol;
        if (gc < N) {
            const __bf16* p = B + (size_t)(k0 + brow) * N + gc;
            rb = *(const v8bf*)p;
            __builtin_prefetch((const void*)(p + (size_t)BK * N), 0, 1); // global_prefetch_b8
        } else {
            rb = v8bf_zero();
        }
    };
    auto storeB = [&](int buf) {
        #pragma unroll
        for (int j = 0; j < 8; ++j)               // store B transposed: [n][k]
            sB[buf][(bcol + j) * LDB_S + brow] = rb[j];
    };

    v8f acc[2][2];
    #pragma unroll
    for (int a = 0; a < 2; ++a)
        #pragma unroll
        for (int b = 0; b < 2; ++b)
            #pragma unroll
            for (int r = 0; r < 8; ++r) acc[a][b][r] = 0.f;

    auto compute = [&](int buf) {
        v16bf af[2], bfv[2];
        #pragma unroll
        for (int mt = 0; mt < 2; ++mt) {
            // A 16x32 frag: lane-half 0 -> K 0..7 & 16..23, half 1 -> 8..15 & 24..31
            const __bf16* base = &sA[buf][(wm * 32 + mt * 16 + l15) * LDA_S + lh * 8];
            v8bf lo = *(const v8bf*)base;
            v8bf hi = *(const v8bf*)(base + 16);
            af[mt] = cat8(lo, hi);
        }
        #pragma unroll
        for (int nt = 0; nt < 2; ++nt) {
            // B 32x16 frag: lane n = l15, lane-half 0 -> K 0..7 & 16..23, half 1 -> 8..15 & 24..31 from Bt LDS
            const __bf16* base = &sB[buf][(wn * 32 + nt * 16 + l15) * LDB_S + lh * 8];
            v8bf lo = *(const v8bf*)base;
            v8bf hi = *(const v8bf*)(base + 16);
            bfv[nt] = cat8(lo, hi);
        }
        #pragma unroll
        for (int mt = 0; mt < 2; ++mt)
            #pragma unroll
            for (int nt = 0; nt < 2; ++nt)
                acc[mt][nt] = __builtin_amdgcn_wmma_f32_16x16x32_bf16(
                    false, af[mt], false, bfv[nt], (short)0, acc[mt][nt], false, false);
    };

    const int KT = K / BK;
    asyncA(0, 0);
    stageB(0);
    waitAsync();
    storeB(0);
    __syncthreads();
    for (int kt = 0; kt < KT; ++kt) {
        int buf = kt & 1, nbuf = buf ^ 1;
        if (kt + 1 < KT) {
            asyncA((kt + 1) * BK, nbuf);   // safe: nbuf last read in compute(kt-1),
            stageB((kt + 1) * BK);         // separated by previous iteration's barrier
        }
        compute(buf);
        if (kt + 1 < KT) {
            waitAsync();                   // own async writes into nbuf complete
            storeB(nbuf);
            __syncthreads();               // single barrier per K-step
        }
    }

    // epilogue: C/D layout -> lane l, VGPR r : row = r + (l>>4)*8, col = l&15
    #pragma unroll
    for (int mt = 0; mt < 2; ++mt) {
        #pragma unroll
        for (int nt = 0; nt < 2; ++nt) {
            int col  = tileN + wn * 32 + nt * 16 + l15;
            int rowb = tileM + wm * 32 + mt * 16 + lh * 8;
            if (col < N) {
                float bv = bias ? bias[col] : 0.f;
                #pragma unroll
                for (int r = 0; r < 8; ++r) {
                    int row = rowb + r;
                    if (row < M) {
                        size_t idx = (size_t)row * N + col;
                        float v = acc[mt][nt][r] + bv;
                        if (add0) v += add0[idx];
                        if (add1) v += add1[idx];
                        if (doRelu) v = fmaxf(v, 0.f);
                        D[idx] = v;
                    }
                }
            }
        }
    }
}

// ---------------------------------------------------------------------------
// Host-side orchestration
// ---------------------------------------------------------------------------
extern "C" void kernel_launch(void* const* d_in, const int* in_sizes, int n_in,
                              void* d_out, int out_size, void* d_ws, size_t ws_size,
                              hipStream_t stream)
{
    const int Nn = GN, E = GE, INF = GINF, HID = GHID, HEADS = GHEADS, NCLS = GNCLS;
    const int FG = HEADS * HID; // 2048

    const float* inputs = (const float*)d_in[0];
    const int*   esrc   = (const int*)  d_in[1];
    const int*   edst   = (const int*)  d_in[2];
    const float* W1s    = (const float*)d_in[3];
    const float* W1n    = (const float*)d_in[4];
    const float* b1     = (const float*)d_in[5];
    const float* W2s    = (const float*)d_in[6];
    const float* W2n    = (const float*)d_in[7];
    const float* b2     = (const float*)d_in[8];
    const float* Wg     = (const float*)d_in[9];
    const float* atl    = (const float*)d_in[10];
    const float* atr    = (const float*)d_in[11];
    const float* Wae    = (const float*)d_in[12];
    const float* Wc     = (const float*)d_in[13];
    const float* bc     = (const float*)d_in[14];
    float* out = (float*)d_out;

    // bump allocator on workspace
    size_t off = 0;
    auto alloc = [&](size_t bytes) -> char* {
        off = (off + 255) & ~(size_t)255;
        char* p = (char*)d_ws + off;
        off += bytes;
        return p;
    };
    __bf16* Xbf   = (__bf16*)alloc((size_t)Nn * INF * 2);
    __bf16* W1sB  = (__bf16*)alloc((size_t)INF * HID * 2);
    __bf16* W1nB  = (__bf16*)alloc((size_t)INF * HID * 2);
    __bf16* W2sB  = (__bf16*)alloc((size_t)HID * HID * 2);
    __bf16* W2nB  = (__bf16*)alloc((size_t)HID * HID * 2);
    __bf16* WgB   = (__bf16*)alloc((size_t)HID * FG  * 2);
    __bf16* WaeB  = (__bf16*)alloc((size_t)INF * HID * 2);
    __bf16* WcB   = (__bf16*)alloc((size_t)FG  * NCLS * 2);
    float*  deg   = (float*) alloc((size_t)Nn * 4);
    float*  zb    = (float*) alloc((size_t)Nn * HID * 4);
    float*  agg   = (float*) alloc((size_t)Nn * INF * 4);     // reused (512 then 256)
    __bf16* aggB  = (__bf16*)alloc((size_t)Nn * INF * 2);     // reused
    float*  hb    = (float*) alloc((size_t)Nn * HID * 4);
    __bf16* hB    = (__bf16*)alloc((size_t)Nn * HID * 2);
    float*  elb   = (float*) alloc((size_t)Nn * HEADS * 4);
    float*  erb   = (float*) alloc((size_t)Nn * HEADS * 4);
    unsigned* emax= (unsigned*)alloc((size_t)Nn * HEADS * 4);
    float*  denom = (float*) alloc((size_t)Nn * HEADS * 4);
    float*  exb   = (float*) alloc((size_t)(E + Nn) * HEADS * 4);
    float*  feat  = (float*) alloc((size_t)Nn * FG * 4);
    float*  gat   = (float*) alloc((size_t)Nn * FG * 4);
    __bf16* gatB  = (__bf16*)alloc((size_t)Nn * FG * 2);
    if (off > ws_size) return;   // insufficient scratch

    const int T = 256;
    auto cast = [&](const float* s, __bf16* d, long n) {
        k_cast_bf16<<<CEILDIV(n, (long)T), T, 0, stream>>>(s, d, n);
    };
    auto gemm = [&](const __bf16* A, const __bf16* B, float* D,
                    const float* a0, const float* a1, const float* bs,
                    int M, int N2, int K, int relu) {
        dim3 g(CEILDIV(M, BM), CEILDIV(N2, BN));
        k_gemm_bf16<<<g, T, 0, stream>>>(A, B, D, a0, a1, bs, M, N2, K, relu);
    };

    // degrees (self loop counted as the initial 1.0)
    k_fill_f32<<<CEILDIV((long)Nn, (long)T), T, 0, stream>>>(deg, 1.0f, Nn);
    k_degree<<<CEILDIV((long)E, (long)T), T, 0, stream>>>(edst, deg, E);

    // bf16 casts of inputs + weights
    cast(inputs, Xbf, (long)Nn * INF);
    cast(W1s, W1sB, (long)INF * HID);
    cast(W1n, W1nB, (long)INF * HID);
    cast(W2s, W2sB, (long)HID * HID);
    cast(W2n, W2nB, (long)HID * HID);
    cast(Wg,  WgB,  (long)HID * FG);
    cast(Wae, WaeB, (long)INF * HID);
    cast(Wc,  WcB,  (long)FG * NCLS);

    // z = X @ W_ae
    gemm(Xbf, WaeB, zb, nullptr, nullptr, nullptr, Nn, HID, INF, 0);

    // --- SAGE layer 1 ---
    hipMemcpyAsync(agg, inputs, (size_t)Nn * INF * 4, hipMemcpyDeviceToDevice, stream);
    k_scatter_add<<<CEILDIV((long)E * (INF / 4), (long)T), T, 0, stream>>>(inputs, esrc, edst, agg, E, INF);
    k_div_deg<<<CEILDIV((long)Nn * INF, (long)T), T, 0, stream>>>(agg, deg, (long)Nn * INF, INF);
    cast(agg, aggB, (long)Nn * INF);
    gemm(Xbf, W1sB, hb, nullptr, nullptr, nullptr, Nn, HID, INF, 0);                 // P = X@W1s
    gemm(aggB, W1nB, hb, hb, zb, b1, Nn, HID, INF, 1);                               // h1 = relu(agg@W1n + P + b1 + z)
    cast(hb, hB, (long)Nn * HID);

    // --- SAGE layer 2 ---
    hipMemcpyAsync(agg, hb, (size_t)Nn * HID * 4, hipMemcpyDeviceToDevice, stream);
    k_scatter_add<<<CEILDIV((long)E * (HID / 4), (long)T), T, 0, stream>>>(hb, esrc, edst, agg, E, HID);
    k_div_deg<<<CEILDIV((long)Nn * HID, (long)T), T, 0, stream>>>(agg, deg, (long)Nn * HID, HID);
    cast(agg, aggB, (long)Nn * HID);
    gemm(hB, W2sB, hb, nullptr, nullptr, nullptr, Nn, HID, HID, 0);                  // P2 = h1@W2s
    gemm(aggB, W2nB, hb, hb, zb, b2, Nn, HID, HID, 1);                               // h2 = relu(agg@W2n + P2 + b2 + z)
    cast(hb, hB, (long)Nn * HID);

    // --- GAT ---
    gemm(hB, WgB, feat, nullptr, nullptr, nullptr, Nn, FG, HID, 0);                  // feat = h2@Wg
    k_el_er<<<Nn, T, 0, stream>>>(feat, atl, atr, elb, erb);
    k_fill_u32<<<CEILDIV((long)Nn * HEADS, (long)T), T, 0, stream>>>(emax, 0x007FFFFFu /*enc(-inf)*/, (long)Nn * HEADS);
    hipMemsetAsync(denom, 0, (size_t)Nn * HEADS * 4, stream);
    hipMemsetAsync(gat,   0, (size_t)Nn * FG * 4, stream);
    long ET = (long)E + Nn;
    k_edge_max<<<CEILDIV(ET * HEADS, (long)T), T, 0, stream>>>(esrc, edst, elb, erb, emax, E, Nn);
    k_edge_exp<<<CEILDIV(ET * HEADS, (long)T), T, 0, stream>>>(esrc, edst, elb, erb, emax, denom, exb, E, Nn);
    k_edge_msg<<<CEILDIV(ET * 128, (long)T), T, 0, stream>>>(esrc, edst, feat, exb, denom, gat, E, Nn);

    // --- classifier ---
    cast(gat, gatB, (long)Nn * FG);
    gemm(gatB, WcB, out, nullptr, nullptr, bc, Nn, NCLS, FG, 0);
}
